// AdjacencyGenerator_5446018531891
// MI455X (gfx1250) — compile-verified, matrix operand order checked by reading, not run on hardware
//
#include <hip/hip_runtime.h>
#include <hip/hip_bf16.h>
#include <math.h>

// ---------------------------------------------------------------------------
// Fused "edge MLP" for the AdjacencyGenerator reference.
// Key identity: softmax over axis=1 sums to 1, so einsum('ijk,il->ikl')
// collapses to h_prime[i,k,l] = Wh[i,l]. The whole attention block cancels.
// Remaining work: per-edge gather + 4 small GEMMs + 3 LayerNorms + final dot,
// then broadcast each edge scalar to D output slots.
// One fused kernel, WMMA f32<-f16 16x16x32, everything staged in LDS.
// ---------------------------------------------------------------------------

typedef __attribute__((ext_vector_type(16))) _Float16 v16h;
typedef __attribute__((ext_vector_type(8)))  float    v8f;

#define DMODEL   128
#define E_EDGES  1024
#define ROWS     16      // edge rows per workgroup (one WMMA M-tile)
#define NTHREADS 128     // 4 waves of 32

__device__ __forceinline__ v8f wmma16(v16h a, v16h b, v8f c) {
  // D = A(16x32 f16) * B(32x16 f16) + C(16x16 f32)
  return __builtin_amdgcn_wmma_f32_16x16x32_f16(
      /*neg_a=*/false, a, /*neg_b=*/false, b,
      /*c_mod=*/(short)0, c, /*reuse_a=*/false, /*reuse_b=*/false);
}

// A fragment (16x32, MxK) from an f32 row-major tile with row stride `ld`.
// ISA 7.12.2 wave32 layout: lane&15 = M; lanes<16 hold K {0..7,16..23},
// lanes>=16 hold K {8..15,24..31}; 2 halves per VGPR.
__device__ __forceinline__ v16h load_A(const float* src, int ld, int kOff, int lane) {
  const int m     = lane & 15;
  const int kBase = (lane >> 4) << 3;
  v16h a;
#pragma unroll
  for (int h = 0; h < 16; ++h) {
    const int K = ((h >> 3) << 4) + kBase + (h & 7);
    a[h] = (_Float16)src[m * ld + kOff + K];
  }
  return a;
}

// B fragment (32x16, KxN), weight stored k-major: B[k][n] = src[(kOff+K)*ld + nOff+n]
__device__ __forceinline__ v16h load_B_kn(const float* __restrict__ src, int ld,
                                          int kOff, int nOff, int lane) {
  const int n     = lane & 15;
  const int kBase = (lane >> 4) << 3;
  v16h b;
#pragma unroll
  for (int h = 0; h < 16; ++h) {
    const int K = ((h >> 3) << 4) + kBase + (h & 7);
    b[h] = (_Float16)src[(kOff + K) * ld + nOff + n];
  }
  return b;
}

// B fragment for x @ w^T with w stored [N][K] row-major: B[k][n] = src[(nOff+n)*ld + kOff+K]
// (per-lane contiguous 8-float runs -> vectorizable global loads)
__device__ __forceinline__ v16h load_B_nk(const float* __restrict__ src, int ld,
                                          int kOff, int nOff, int lane) {
  const int n     = lane & 15;
  const int kBase = (lane >> 4) << 3;
  v16h b;
#pragma unroll
  for (int h = 0; h < 16; ++h) {
    const int K = ((h >> 3) << 4) + kBase + (h & 7);
    b[h] = (_Float16)src[(nOff + n) * ld + kOff + K];
  }
  return b;
}

// C/D fragment scatter: lane&15 = N, VGPR r -> M = (lane>>4)*8 + r
__device__ __forceinline__ void store_C(float* dst, int ld, int nOff, int lane, v8f c) {
  const int n     = lane & 15;
  const int mBase = (lane >> 4) << 3;
#pragma unroll
  for (int r = 0; r < 8; ++r)
    dst[(mBase + r) * ld + nOff + n] = c[r];
}

__global__ __launch_bounds__(NTHREADS) void fused_edge_mlp(
    const int*   __restrict__ edge_index,   // [2, E]
    const float* __restrict__ x,            // [N, D]
    const float* __restrict__ W,            // [D, D]    (q @ W)
    const float* __restrict__ ff_w,         // [D, D]    (q @ ff_w^T)
    const float* __restrict__ ff_b,         // [D]
    const float* __restrict__ na_g, const float* __restrict__ na_b,
    const float* __restrict__ nf_g, const float* __restrict__ nf_b,
    const float* __restrict__ wl_w,         // [3D, D]
    const float* __restrict__ wl_b,         // [3D]
    const float* __restrict__ w5_w,         // [D, 3D]
    const float* __restrict__ w5_b,         // [D]
    const float* __restrict__ fn_g, const float* __restrict__ fn_b,
    const float* __restrict__ wv_w,         // [D] (row of [1,D])
    const float* __restrict__ wv_b,         // [1]
    float*       __restrict__ out)          // [E*D]
{
  __shared__ float sAct[ROWS][DMODEL];      // current activation (A operand)
  __shared__ float sTmp[ROWS][DMODEL];      // GEMM output staging
  __shared__ float sQ4 [ROWS][DMODEL];      // residual copy (q4)
  __shared__ float sY1 [ROWS][3 * DMODEL];  // hidden 3D activation
  __shared__ float sMean[ROWS], sRstd[ROWS], sScalar[ROWS];

  const int tid       = threadIdx.x;
  const int lane      = tid & 31;
  const int wave      = tid >> 5;
  const int blockBase = blockIdx.x * ROWS;

  // --- prefetch the weight working set (global_prefetch_b8 path) -----------
#pragma unroll 1
  for (int line = tid; line < (DMODEL * DMODEL) / 32; line += NTHREADS) {
    __builtin_prefetch(W    + line * 32, 0, 1);
    __builtin_prefetch(ff_w + line * 32, 0, 1);
  }
#pragma unroll 1
  for (int line = tid; line < (3 * DMODEL * DMODEL) / 32; line += NTHREADS) {
    __builtin_prefetch(wl_w + line * 32, 0, 1);
    __builtin_prefetch(w5_w + line * 32, 0, 1);
  }

  // --- gather x_j = x[edge_index[1]] for this row tile ---------------------
  for (int idx = tid; idx < ROWS * DMODEL; idx += NTHREADS) {
    const int r = idx >> 7, c = idx & (DMODEL - 1);
    const int j = edge_index[E_EDGES + blockBase + r];
    sAct[r][c] = x[j * DMODEL + c];
  }
  __syncthreads();

  // --- GEMM1: Wh = x_j @ W  -> sTmp ----------------------------------------
  for (int nt = wave; nt < 8; nt += 4) {
    v8f c = {};
#pragma unroll
    for (int kt = 0; kt < 4; ++kt) {
      v16h A = load_A(&sAct[0][0], DMODEL, kt * 32, lane);
      v16h B = load_B_kn(W, DMODEL, kt * 32, nt * 16, lane);
      c = wmma16(A, B, c);
    }
    store_C(&sTmp[0][0], DMODEL, nt * 16, lane, c);
  }
  __syncthreads();

  // --- elu -> sAct ---------------------------------------------------------
  for (int idx = tid; idx < ROWS * DMODEL; idx += NTHREADS) {
    const float v = (&sTmp[0][0])[idx];
    (&sAct[0][0])[idx] = (v > 0.f) ? v : (__expf(v) - 1.f);
  }
  __syncthreads();

  // --- LayerNorm (na) in place ---------------------------------------------
  if (tid < ROWS) {
    float s = 0.f, ss = 0.f;
    for (int c = 0; c < DMODEL; ++c) { const float v = sAct[tid][c]; s += v; ss += v * v; }
    const float m = s * (1.f / DMODEL);
    sMean[tid] = m;
    sRstd[tid] = rsqrtf(ss * (1.f / DMODEL) - m * m + 1e-5f);
  }
  __syncthreads();
  for (int idx = tid; idx < ROWS * DMODEL; idx += NTHREADS) {
    const int r = idx >> 7, c = idx & (DMODEL - 1);
    sAct[r][c] = (sAct[r][c] - sMean[r]) * sRstd[r] * na_g[c] + na_b[c];
  }
  __syncthreads();

  // --- GEMM2: t = q @ ff_w^T -> sTmp ---------------------------------------
  for (int nt = wave; nt < 8; nt += 4) {
    v8f c = {};
#pragma unroll
    for (int kt = 0; kt < 4; ++kt) {
      v16h A = load_A(&sAct[0][0], DMODEL, kt * 32, lane);
      v16h B = load_B_nk(ff_w, DMODEL, kt * 32, nt * 16, lane);
      c = wmma16(A, B, c);
    }
    store_C(&sTmp[0][0], DMODEL, nt * 16, lane, c);
  }
  __syncthreads();

  // --- + ff_b, leaky(0.2) -> sAct ------------------------------------------
  for (int idx = tid; idx < ROWS * DMODEL; idx += NTHREADS) {
    const int r = idx >> 7, c = idx & (DMODEL - 1);
    float v = sTmp[r][c] + ff_b[c];
    sAct[r][c] = (v >= 0.f) ? v : 0.2f * v;
  }
  __syncthreads();

  // --- LayerNorm (nf) -> sAct (=q4), keep copy in sQ4 ----------------------
  if (tid < ROWS) {
    float s = 0.f, ss = 0.f;
    for (int c = 0; c < DMODEL; ++c) { const float v = sAct[tid][c]; s += v; ss += v * v; }
    const float m = s * (1.f / DMODEL);
    sMean[tid] = m;
    sRstd[tid] = rsqrtf(ss * (1.f / DMODEL) - m * m + 1e-5f);
  }
  __syncthreads();
  for (int idx = tid; idx < ROWS * DMODEL; idx += NTHREADS) {
    const int r = idx >> 7, c = idx & (DMODEL - 1);
    const float v = (sAct[r][c] - sMean[r]) * sRstd[r] * nf_g[c] + nf_b[c];
    sAct[r][c] = v;
    sQ4[r][c]  = v;
  }
  __syncthreads();

  // --- GEMM3: y1 = q4 @ wl_w^T -> sY1 (16 x 384) ---------------------------
  for (int nt = wave; nt < 24; nt += 4) {
    v8f c = {};
#pragma unroll
    for (int kt = 0; kt < 4; ++kt) {
      v16h A = load_A(&sAct[0][0], DMODEL, kt * 32, lane);
      v16h B = load_B_nk(wl_w, DMODEL, kt * 32, nt * 16, lane);
      c = wmma16(A, B, c);
    }
    store_C(&sY1[0][0], 3 * DMODEL, nt * 16, lane, c);
  }
  __syncthreads();

  // --- + wl_b, leaky(0.2) --------------------------------------------------
  for (int idx = tid; idx < ROWS * 3 * DMODEL; idx += NTHREADS) {
    const int r = idx / (3 * DMODEL), c = idx - r * (3 * DMODEL);
    float v = sY1[r][c] + wl_b[c];
    sY1[r][c] = (v >= 0.f) ? v : 0.2f * v;
  }
  __syncthreads();

  // --- GEMM4: y2 = y1 @ w5_w^T (K = 384) -> sTmp ---------------------------
  for (int nt = wave; nt < 8; nt += 4) {
    v8f c = {};
#pragma unroll
    for (int kt = 0; kt < 12; ++kt) {
      v16h A = load_A(&sY1[0][0], 3 * DMODEL, kt * 32, lane);
      v16h B = load_B_nk(w5_w, 3 * DMODEL, kt * 32, nt * 16, lane);
      c = wmma16(A, B, c);
    }
    store_C(&sTmp[0][0], DMODEL, nt * 16, lane, c);
  }
  __syncthreads();

  // --- y3 = y2 + w5_b + q4 -> sAct ------------------------------------------
  for (int idx = tid; idx < ROWS * DMODEL; idx += NTHREADS) {
    const int r = idx >> 7, c = idx & (DMODEL - 1);
    sAct[r][c] = sTmp[r][c] + w5_b[c] + sQ4[r][c];
  }
  __syncthreads();

  // --- LayerNorm (fn) ------------------------------------------------------
  if (tid < ROWS) {
    float s = 0.f, ss = 0.f;
    for (int c = 0; c < DMODEL; ++c) { const float v = sAct[tid][c]; s += v; ss += v * v; }
    const float m = s * (1.f / DMODEL);
    sMean[tid] = m;
    sRstd[tid] = rsqrtf(ss * (1.f / DMODEL) - m * m + 1e-5f);
  }
  __syncthreads();
  for (int idx = tid; idx < ROWS * DMODEL; idx += NTHREADS) {
    const int r = idx >> 7, c = idx & (DMODEL - 1);
    sAct[r][c] = (sAct[r][c] - sMean[r]) * sRstd[r] * fn_g[c] + fn_b[c];
  }
  __syncthreads();

  // --- final dot with wv_w + wv_b: one scalar per edge row -----------------
  if (tid < ROWS) {
    float s = wv_b[0];
    for (int c = 0; c < DMODEL; ++c) s += sAct[tid][c] * wv_w[c];
    sScalar[tid] = s;
  }
  __syncthreads();

  // --- broadcast each edge scalar to its D replicated output rows ----------
  for (int idx = tid; idx < ROWS * DMODEL; idx += NTHREADS) {
    const int r = idx >> 7, c = idx & (DMODEL - 1);
    out[(blockBase + r) * DMODEL + c] = sScalar[r];
  }
}

extern "C" void kernel_launch(void* const* d_in, const int* in_sizes, int n_in,
                              void* d_out, int out_size, void* d_ws, size_t ws_size,
                              hipStream_t stream) {
  (void)in_sizes; (void)n_in; (void)d_ws; (void)ws_size; (void)out_size;
  const int*   edge_index = (const int*)  d_in[0];
  const float* x_nodes    = (const float*)d_in[1];
  const float* W_mat      = (const float*)d_in[2];
  // d_in[3] = 'a' (attention vector) — algebraically cancels (softmax sums to 1)
  const float* ff_w = (const float*)d_in[4];
  const float* ff_b = (const float*)d_in[5];
  const float* na_g = (const float*)d_in[6];
  const float* na_b = (const float*)d_in[7];
  const float* nf_g = (const float*)d_in[8];
  const float* nf_b = (const float*)d_in[9];
  const float* wl_w = (const float*)d_in[10];
  const float* wl_b = (const float*)d_in[11];
  const float* w5_w = (const float*)d_in[12];
  const float* w5_b = (const float*)d_in[13];
  const float* fn_g = (const float*)d_in[14];
  const float* fn_b = (const float*)d_in[15];
  const float* wv_w = (const float*)d_in[16];
  const float* wv_b = (const float*)d_in[17];
  float* out = (float*)d_out;

  dim3 grid(E_EDGES / ROWS);   // 64 workgroups
  dim3 block(NTHREADS);        // 4 wave32s
  hipLaunchKernelGGL(fused_edge_mlp, grid, block, 0, stream,
                     edge_index, x_nodes, W_mat, ff_w, ff_b,
                     na_g, na_b, nf_g, nf_b,
                     wl_w, wl_b, w5_w, w5_b,
                     fn_g, fn_b, wv_w, wv_b, out);
}
